// RDurendal_29231547417242
// MI455X (gfx1250) — matrix operand order repaired and checked by reading, not compiled, not run on hardware
//
#include <hip/hip_runtime.h>

// ---------------------------------------------------------------------------
// Problem constants (match reference)
// ---------------------------------------------------------------------------
static constexpr int NN = 100000;   // nodes
static constexpr int RR = 3;        // relations
static constexpr int EE = 800000;   // edges per relation
static constexpr int LL = 200000;   // label edges per relation
static constexpr int HH = 64;       // FIN == H1 == H2 == 64
static constexpr int NTILES = NN / 16;   // 6250 exact

typedef __attribute__((ext_vector_type(16))) _Float16 v16h;
typedef __attribute__((ext_vector_type(8)))  float    v8f;

// ---------------------------------------------------------------------------
// WMMA helpers (CDNA5 v_wmma_f32_16x16x32_f16, wave32)
// ---------------------------------------------------------------------------
__device__ __forceinline__ v8f wmma_f16(v16h a, v16h b, v8f c) {
  return __builtin_amdgcn_wmma_f32_16x16x32_f16(
      /*neg_a=*/false, a, /*neg_b=*/false, b,
      /*c_mod=*/(short)0, c, /*reuse_a=*/false, /*reuse_b=*/false);
}

// A-fragment (16x32, f16) from an f32 row-major matrix (ld floats per row).
// ISA layout: lanes 0-15 row M=lane, elems 0-7 => K=kb+0..7, 8-15 => K=kb+16..23
//             lanes 16-31 row M=lane-16, elems 0-7 => K=kb+8..15, 8-15 => K=kb+24..31
__device__ __forceinline__ v16h load_A_f32(const float* __restrict__ base,
                                           int ld, int kb, int lane) {
  const int half = (lane >> 4) & 1;
  const int m    = lane & 15;
  const float* p0 = base + (size_t)m * ld + kb + half * 8;
  const float* p1 = base + (size_t)m * ld + kb + 16 + half * 8;
  v16h a;
#pragma unroll
  for (int i = 0; i < 8; ++i) a[i]     = (_Float16)p0[i];
#pragma unroll
  for (int i = 0; i < 8; ++i) a[8 + i] = (_Float16)p1[i];
  return a;
}

// Same A-fragment from an f16 row-major matrix (ld halves per row).
__device__ __forceinline__ v16h load_A_h16(const _Float16* __restrict__ base,
                                           int ld, int kb, int lane) {
  const int half = (lane >> 4) & 1;
  const int m    = lane & 15;
  const _Float16* p0 = base + (size_t)m * ld + kb + half * 8;
  const _Float16* p1 = base + (size_t)m * ld + kb + 16 + half * 8;
  v16h a;
#pragma unroll
  for (int i = 0; i < 8; ++i) a[i]     = p0[i];
#pragma unroll
  for (int i = 0; i < 8; ++i) a[8 + i] = p1[i];
  return a;
}

// B-fragment (32x16) computing D = A * W^T : B[k][n] = W[nt*16+n][kb + ...].
// ISA layout: col N=lane%16, elems 0-7 => K=kb+half*8+0..7, 8-15 => K=kb+16+half*8+0..7.
__device__ __forceinline__ v16h load_B_h16(const _Float16* __restrict__ W,
                                           int ldw, int nt, int kb, int lane) {
  const int half = (lane >> 4) & 1;
  const int n    = lane & 15;
  const _Float16* p = W + (size_t)(nt * 16 + n) * ldw + kb + half * 8;
  v16h b;
#pragma unroll
  for (int i = 0; i < 8; ++i) b[i] = p[i];
#pragma unroll
  for (int i = 0; i < 8; ++i) b[8 + i] = p[16 + i];
  return b;
}

__device__ __forceinline__ float fsigmoid(float x) {
  return 1.0f / (1.0f + __expf(-x));
}

// ---------------------------------------------------------------------------
// Utility kernels
// ---------------------------------------------------------------------------
__global__ void zero4_kernel(float4* __restrict__ p, long n4) {
  long i = (long)blockIdx.x * blockDim.x + threadIdx.x;
  if (i < n4) p[i] = make_float4(0.f, 0.f, 0.f, 0.f);
}

__global__ void cvt_f16_kernel(const float* __restrict__ s,
                               _Float16* __restrict__ d, int n) {
  int i = blockIdx.x * blockDim.x + threadIdx.x;
  if (i < n) d[i] = (_Float16)s[i];
}

// ---------------------------------------------------------------------------
// Scatter-add: agg[r][dst] += x[src]   (segment_sum).  grid.y = relation.
// One thread handles 16 channels of one edge.
// ---------------------------------------------------------------------------
__global__ void scatter_add_kernel(const float* __restrict__ x,
                                   const int* __restrict__ ei,
                                   float* __restrict__ agg) {
  long idx = (long)blockIdx.x * blockDim.x + threadIdx.x;
  const int r = blockIdx.y;
  if (idx >= (long)EE * 4) return;
  const int e  = (int)(idx >> 2);
  const int cg = (int)(idx & 3) * 16;
  const int src = ei[(size_t)r * 2 * EE + e];
  const int dst = ei[(size_t)r * 2 * EE + EE + e];
  const float* xs = x + (size_t)src * HH + cg;
  float* ag = agg + ((size_t)r * NN + dst) * HH + cg;
#pragma unroll
  for (int i = 0; i < 16; i += 4) {
    float4 v = *(const float4*)(xs + i);
    atomicAdd(ag + i + 0, v.x);
    atomicAdd(ag + i + 1, v.y);
    atomicAdd(ag + i + 2, v.z);
    atomicAdd(ag + i + 3, v.w);
  }
}

// ---------------------------------------------------------------------------
// Fused graph-conv + GRU per (relation, 16-node tile).  One wave per tile.
// h  = relu(agg @ Wrel^T + brel + xin @ Wroot^T)                (16 WMMA)
// gi = h @ Wi^T + bi ; gh = past @ Wh^T + bh                    (48 WMMA)
// cur = (1-z)*tanh(inn + r*hn) + z*past
// ---------------------------------------------------------------------------
__global__ __launch_bounds__(128)
void dense_gru_kernel(const float* __restrict__ agg,       // [R][N][64]
                      const float* __restrict__ xin,       // [N][64]
                      const _Float16* __restrict__ Wrel,   // [R][64][64]
                      const _Float16* __restrict__ Wroot,  // [R][64][64]
                      const float* __restrict__ brel,      // [R][64]
                      const _Float16* __restrict__ Wi,     // [192][64]
                      const _Float16* __restrict__ Wh,     // [192][64]
                      const float* __restrict__ bi,        // [192]
                      const float* __restrict__ bh,        // [192]
                      const float* __restrict__ past,      // [R][N][64]
                      float* __restrict__ cur)             // [R][N][64]
{
  __shared__ _Float16 hst[4][16][HH];   // per-wave h staging (C->A re-stripe)
  const int lane = threadIdx.x & 31;
  const int wave = threadIdx.x >> 5;
  const int r    = blockIdx.y;
  const int tile = blockIdx.x * 4 + wave;
  if (tile >= NTILES) return;           // no block barriers used: safe
  const int row0 = tile * 16;

  const float*    aggR   = agg  + ((size_t)r * NN + row0) * HH;
  const float*    xR     = xin  + (size_t)row0 * HH;
  const float*    pastR  = past + ((size_t)r * NN + row0) * HH;
  float*          curR   = cur  + ((size_t)r * NN + row0) * HH;
  const _Float16* WrelR  = Wrel  + (size_t)r * HH * HH;
  const _Float16* WrootR = Wroot + (size_t)r * HH * HH;
  const float*    brelR  = brel  + r * HH;
  const int half = lane >> 4, n = lane & 15;

  // ---- graph conv -------------------------------------------------------
  v8f acc[4] = {};
#pragma unroll
  for (int kb = 0; kb < HH; kb += 32) {
    v16h aA = load_A_f32(aggR, HH, kb, lane);
    v16h aX = load_A_f32(xR,   HH, kb, lane);
#pragma unroll
    for (int t = 0; t < 4; ++t) {
      acc[t] = wmma_f16(aA, load_B_h16(WrelR,  HH, t, kb, lane), acc[t]);
      acc[t] = wmma_f16(aX, load_B_h16(WrootR, HH, t, kb, lane), acc[t]);
    }
  }
  // bias + relu, stash to LDS as f16 in row-major [16][64]
#pragma unroll
  for (int t = 0; t < 4; ++t) {
    const float bv = brelR[t * 16 + n];
#pragma unroll
    for (int v = 0; v < 8; ++v) {
      float hv = acc[t][v] + bv;
      hv = hv > 0.f ? hv : 0.f;
      hst[wave][v + 8 * half][t * 16 + n] = (_Float16)hv;
    }
  }
  asm volatile("s_wait_dscnt 0" ::: "memory");   // store->load hazard, same wave
  const _Float16* hrows = &hst[wave][0][0];

  // ---- GRU gates --------------------------------------------------------
  v8f rz[8]  = {};   // [0..3]=r-gate (gi+gh), [4..7]=z-gate (gi+gh)
  v8f nin[4] = {};   // inn
  v8f nhh[4] = {};   // hn
#pragma unroll
  for (int kb = 0; kb < HH; kb += 32) {
    v16h aH = load_A_h16(hrows, HH, kb, lane);
    v16h aP = load_A_f32(pastR, HH, kb, lane);
#pragma unroll
    for (int t = 0; t < 8; ++t) {
      rz[t] = wmma_f16(aH, load_B_h16(Wi, HH, t, kb, lane), rz[t]);
      rz[t] = wmma_f16(aP, load_B_h16(Wh, HH, t, kb, lane), rz[t]);
    }
#pragma unroll
    for (int t = 0; t < 4; ++t) {
      nin[t] = wmma_f16(aH, load_B_h16(Wi, HH, 8 + t, kb, lane), nin[t]);
      nhh[t] = wmma_f16(aP, load_B_h16(Wh, HH, 8 + t, kb, lane), nhh[t]);
    }
  }

  // ---- elementwise GRU --------------------------------------------------
#pragma unroll
  for (int t = 0; t < 4; ++t) {
    const int   c    = t * 16 + n;
    const float birh = bi[c] + bh[c];
    const float bizh = bi[HH + c] + bh[HH + c];
    const float bin_ = bi[2 * HH + c];
    const float bhn_ = bh[2 * HH + c];
#pragma unroll
    for (int v = 0; v < 8; ++v) {
      const int m = v + 8 * half;
      const float rg = fsigmoid(rz[t][v] + birh);
      const float zg = fsigmoid(rz[4 + t][v] + bizh);
      const float hn = nhh[t][v] + bhn_;
      const float nn = tanhf(nin[t][v] + bin_ + rg * hn);
      const float pv = pastR[(size_t)m * HH + c];
      curR[(size_t)m * HH + c] = (1.f - zg) * nn + zg * pv;
    }
  }
}

// ---------------------------------------------------------------------------
// Semantic attention: asum[r][f] += sum_rows tanh((cur @ kW^T + kb))[.,f]
// ---------------------------------------------------------------------------
__global__ __launch_bounds__(128)
void attn_partial_kernel(const float* __restrict__ cur,     // [R][N][64]
                         const _Float16* __restrict__ kW,   // [64][64] f16
                         const float* __restrict__ kb,      // [64]
                         float* __restrict__ asum)          // [R][64]
{
  const int lane = threadIdx.x & 31;
  const int wave = threadIdx.x >> 5;
  const int r    = blockIdx.y;
  const int tile = blockIdx.x * 4 + wave;
  if (tile >= NTILES) return;
  const float* curR = cur + ((size_t)r * NN + (size_t)tile * 16) * HH;
  const int half = lane >> 4, n = lane & 15;

  v8f acc[4] = {};
#pragma unroll
  for (int kb0 = 0; kb0 < HH; kb0 += 32) {
    v16h a = load_A_f32(curR, HH, kb0, lane);
#pragma unroll
    for (int t = 0; t < 4; ++t)
      acc[t] = wmma_f16(a, load_B_h16(kW, HH, t, kb0, lane), acc[t]);
  }
#pragma unroll
  for (int t = 0; t < 4; ++t) {
    const float bv = kb[t * 16 + n];
    float s = 0.f;
#pragma unroll
    for (int v = 0; v < 8; ++v) s += tanhf(acc[t][v] + bv);
    s += __shfl_xor(s, 16, 32);              // sum the two row halves
    if (half == 0) atomicAdd(&asum[r * HH + t * 16 + n], s);
  }
}

// score[r] = sum_f q[f]*asum[r][f]/N ; alpha = softmax_r(score)
__global__ void attn_final_kernel(const float* __restrict__ asum,
                                  const float* __restrict__ q,
                                  float* __restrict__ alpha) {
  const int lane = threadIdx.x;   // 32 threads
  float s[RR];
#pragma unroll
  for (int r = 0; r < RR; ++r) {
    float p = q[lane] * asum[r * HH + lane] +
              q[lane + 32] * asum[r * HH + lane + 32];
#pragma unroll
    for (int off = 16; off; off >>= 1) p += __shfl_xor(p, off, 32);
    s[r] = p * (1.0f / (float)NN);
  }
  if (lane == 0) {
    float mx = fmaxf(s[0], fmaxf(s[1], s[2]));
    float e0 = __expf(s[0] - mx), e1 = __expf(s[1] - mx), e2 = __expf(s[2] - mx);
    float inv = 1.f / (e0 + e1 + e2);
    alpha[0] = e0 * inv; alpha[1] = e1 * inv; alpha[2] = e2 * inv;
  }
}

// out[n][c] = sum_r alpha[r] * cur[r][n][c]
__global__ void combine_kernel(const float* __restrict__ cur,
                               const float* __restrict__ alpha,
                               float* __restrict__ out) {
  long i = (long)blockIdx.x * blockDim.x + threadIdx.x;
  if (i < (long)NN * HH) {
    out[i] = alpha[0] * cur[i] +
             alpha[1] * cur[(size_t)NN * HH + i] +
             alpha[2] * cur[2 * (size_t)NN * HH + i];
  }
}

// z[n][k] = agg2[n] . post_W[k] + post_b[k]   (k = 0,1)
__global__ void postz_kernel(const float* __restrict__ comb,
                             const float* __restrict__ pw,
                             const float* __restrict__ pb,
                             float* __restrict__ z) {
  int nidx = blockIdx.x * blockDim.x + threadIdx.x;
  if (nidx >= NN) return;
  const float* row = comb + (size_t)nidx * HH;
  float z0 = pb[0], z1 = pb[1];
#pragma unroll 8
  for (int c = 0; c < HH; ++c) {
    const float v = row[c];
    z0 += v * pw[c];
    z1 += v * pw[HH + c];
  }
  z[nidx * 2]     = z0;
  z[nidx * 2 + 1] = z1;
}

// complex-rotation link scores
__global__ void score_kernel(const float* __restrict__ z,
                             const int* __restrict__ eli,
                             const float* __restrict__ rel,
                             float* __restrict__ out) {
  int l = blockIdx.x * blockDim.x + threadIdx.x;
  const int r = blockIdx.y;
  if (l >= LL) return;
  const int hidx = eli[(size_t)r * 2 * LL + l];
  const int tidx = eli[(size_t)r * 2 * LL + LL + l];
  const float hr = z[hidx * 2],  hi = z[hidx * 2 + 1];
  const float tr = z[tidx * 2],  ti = z[tidx * 2 + 1];
  const float rr = rel[r * 2],   ri = rel[r * 2 + 1];
  out[(size_t)r * LL + l] = hr * rr * tr + hi * rr * ti + hr * ri * ti - hi * ri * tr;
}

// ---------------------------------------------------------------------------
// Host launcher
// ---------------------------------------------------------------------------
extern "C" void kernel_launch(void* const* d_in, const int* in_sizes, int n_in,
                              void* d_out, int out_size, void* d_ws, size_t ws_size,
                              hipStream_t stream) {
  (void)in_sizes; (void)n_in; (void)out_size; (void)ws_size;

  const float* x        = (const float*)d_in[0];
  const int*   ei       = (const int*)  d_in[1];
  const int*   eli      = (const int*)  d_in[2];
  const float* past1    = (const float*)d_in[3];
  const float* past2    = (const float*)d_in[4];
  const float* b1_rel   = (const float*)d_in[6];
  const float* gru1_bi  = (const float*)d_in[10];
  const float* gru1_bh  = (const float*)d_in[11];
  const float* k1_b     = (const float*)d_in[13];
  const float* q1       = (const float*)d_in[14];
  const float* b2_rel   = (const float*)d_in[16];
  const float* gru2_bi  = (const float*)d_in[20];
  const float* gru2_bh  = (const float*)d_in[21];
  const float* k2_b     = (const float*)d_in[23];
  const float* q2       = (const float*)d_in[24];
  const float* post_W   = (const float*)d_in[25];
  const float* post_b   = (const float*)d_in[26];
  const float* rel_emb  = (const float*)d_in[27];

  float* out      = (float*)d_out;
  float* scores   = out;                                   // [R][L]
  float* cur1     = out + (size_t)RR * LL;                 // [R][N][64]
  float* cur2     = cur1 + (size_t)RR * NN * HH;           // [R][N][64]

  // ---- carve workspace --------------------------------------------------
  char* ws = (char*)d_ws;
  size_t off = 0;
  auto carve = [&](size_t bytes) -> void* {
    void* p = ws + off;
    off += (bytes + 255) & ~(size_t)255;
    return p;
  };
  float* aggbuf = (float*)carve((size_t)RR * NN * HH * sizeof(float)); // 76.8MB
  float* comb   = (float*)carve((size_t)NN * HH * sizeof(float));      // 25.6MB
  float* zbuf   = (float*)carve((size_t)NN * 2 * sizeof(float));
  float* asum   = (float*)carve(2 * RR * HH * sizeof(float));
  float* alpha  = (float*)carve(8 * sizeof(float));
  _Float16* w1rel16  = (_Float16*)carve((size_t)RR * HH * HH * 2);
  _Float16* w1root16 = (_Float16*)carve((size_t)RR * HH * HH * 2);
  _Float16* wi1_16   = (_Float16*)carve((size_t)3 * HH * HH * 2);
  _Float16* wh1_16   = (_Float16*)carve((size_t)3 * HH * HH * 2);
  _Float16* k1_16    = (_Float16*)carve((size_t)HH * HH * 2);
  _Float16* w2rel16  = (_Float16*)carve((size_t)RR * HH * HH * 2);
  _Float16* w2root16 = (_Float16*)carve((size_t)RR * HH * HH * 2);
  _Float16* wi2_16   = (_Float16*)carve((size_t)3 * HH * HH * 2);
  _Float16* wh2_16   = (_Float16*)carve((size_t)3 * HH * HH * 2);
  _Float16* k2_16    = (_Float16*)carve((size_t)HH * HH * 2);

  // ---- weight f32 -> f16 ------------------------------------------------
  auto cvt = [&](int which, _Float16* dst, int nElem) {
    cvt_f16_kernel<<<(nElem + 255) / 256, 256, 0, stream>>>(
        (const float*)d_in[which], dst, nElem);
  };
  cvt(5,  w1rel16,  RR * HH * HH);
  cvt(7,  w1root16, RR * HH * HH);
  cvt(8,  wi1_16,   3 * HH * HH);
  cvt(9,  wh1_16,   3 * HH * HH);
  cvt(12, k1_16,    HH * HH);
  cvt(15, w2rel16,  RR * HH * HH);
  cvt(17, w2root16, RR * HH * HH);
  cvt(18, wi2_16,   3 * HH * HH);
  cvt(19, wh2_16,   3 * HH * HH);
  cvt(22, k2_16,    HH * HH);

  const long agg4 = (long)RR * NN * HH / 4;
  const dim3 denseGrid((NTILES + 3) / 4, RR);
  const dim3 scatGrid((EE * 4 + 255) / 256, RR);
  const dim3 scoreGrid((LL + 255) / 256, RR);

  // ---- layer 1 ----------------------------------------------------------
  zero4_kernel<<<(unsigned)((agg4 + 255) / 256), 256, 0, stream>>>((float4*)aggbuf, agg4);
  zero4_kernel<<<1, 256, 0, stream>>>((float4*)asum, (long)(2 * RR * HH) / 4);
  scatter_add_kernel<<<scatGrid, 256, 0, stream>>>(x, ei, aggbuf);
  dense_gru_kernel<<<denseGrid, 128, 0, stream>>>(
      aggbuf, x, w1rel16, w1root16, b1_rel,
      wi1_16, wh1_16, gru1_bi, gru1_bh, past1, cur1);
  attn_partial_kernel<<<denseGrid, 128, 0, stream>>>(cur1, k1_16, k1_b, asum);
  attn_final_kernel<<<1, 32, 0, stream>>>(asum, q1, alpha);
  combine_kernel<<<((long)NN * HH + 255) / 256, 256, 0, stream>>>(cur1, alpha, comb);

  // ---- layer 2 ----------------------------------------------------------
  zero4_kernel<<<(unsigned)((agg4 + 255) / 256), 256, 0, stream>>>((float4*)aggbuf, agg4);
  scatter_add_kernel<<<scatGrid, 256, 0, stream>>>(comb, ei, aggbuf);
  dense_gru_kernel<<<denseGrid, 128, 0, stream>>>(
      aggbuf, comb, w2rel16, w2root16, b2_rel,
      wi2_16, wh2_16, gru2_bi, gru2_bh, past2, cur2);
  attn_partial_kernel<<<denseGrid, 128, 0, stream>>>(cur2, k2_16, k2_b, asum + RR * HH);
  attn_final_kernel<<<1, 32, 0, stream>>>(asum + RR * HH, q2, alpha + 4);
  combine_kernel<<<((long)NN * HH + 255) / 256, 256, 0, stream>>>(cur2, alpha + 4, comb);

  // ---- post projection + link scores ------------------------------------
  postz_kernel<<<(NN + 255) / 256, 256, 0, stream>>>(comb, post_W, post_b, zbuf);
  score_kernel<<<scoreGrid, 256, 0, stream>>>(zbuf, eli, rel_emb, scores);
}
